// MultiHeadAttention_41944650613197
// MI455X (gfx1250) — matrix operand order repaired and checked by reading, not compiled, not run on hardware
//
#include <hip/hip_runtime.h>

// MI455X (gfx1250, CDNA5) causal multi-head attention forward.
// Pipeline:
//   1) convert x_q/x_k/x_v f32 -> bf16 (row-major)
//   2) transpose-convert Wq/Wk/Wv/Wo f32[K][N] -> bf16 [N][K]
//   3) Q/K/V projections: pure-bf16 GEMM (contiguous A and B loads, 32x64/wave)
//      Q,K written [B,H,S,DH]; V written transposed [B,H,DH,S]
//   4) flash attention (WMMA QK^T + online softmax + WMMA P.V, all contiguous)
//   5) output projection -> f32
// All matrix math on v_wmma_f32_16x16x32_bf16 (f32 accumulate).

#define BATCH  4
#define SEQ    2048
#define DMODEL 1024
#define NH     16
#define DHEAD  64
#define ATT_SCALE 0.125f          // 1/sqrt(64)
#define NEG_BIG  (-3.0e38f)

typedef __bf16 bf16_t;
typedef __attribute__((ext_vector_type(16))) __bf16 v16bf;
typedef __attribute__((ext_vector_type(8)))  __bf16 v8bf;
typedef __attribute__((ext_vector_type(8)))  float  v8f;

static __device__ __forceinline__ v8f wmma_bf16(v16bf a, v16bf b, v8f c) {
  // (neg_a, A, neg_b, B, c_mod, C, reuse_a, reuse_b)
  return __builtin_amdgcn_wmma_f32_16x16x32_bf16(false, a, false, b, (short)0, c,
                                                 false, false);
}

// ---- fragment builders (layouts per CDNA5 ISA 7.12.2, wave32) -------------
// A (16x32 bf16): lane m (0..15 per half-wave) = row M=m.
//   elems 0..7  : K = half*8 + e        elems 8..15 : K = 16 + half*8 + (e-8)
static __device__ __forceinline__ v16bf afrag_from_bf16(const bf16_t* rowp, int half) {
  v8bf r0 = *(const v8bf*)(rowp + half * 8);
  v8bf r1 = *(const v8bf*)(rowp + 16 + half * 8);
  v16bf a;
#pragma unroll
  for (int e = 0; e < 8; ++e) { a[e] = r0[e]; a[8 + e] = r1[e]; }
  return a;
}
// B (32x16 bf16): lane = half*16 + n; elems 0..7: K = half*8 + e, elems 8..15: K = 16 + half*8 + (e-8), col N = n.
// rowp points at the n-th row of a [N][K]-major matrix, K contiguous.
static __device__ __forceinline__ v16bf bfrag_rows_contig(const bf16_t* rowp, int half) {
  v8bf r0 = *(const v8bf*)(rowp + half * 8);
  v8bf r1 = *(const v8bf*)(rowp + 16 + half * 8);
  v16bf b;
#pragma unroll
  for (int e = 0; e < 8; ++e) { b[e] = r0[e]; b[8 + e] = r1[e]; }
  return b;
}

static __device__ __forceinline__ float rowmax16(float x) {
#pragma unroll
  for (int off = 1; off < 16; off <<= 1) x = fmaxf(x, __shfl_xor(x, off, 32));
  return x;
}
static __device__ __forceinline__ float rowsum16(float x) {
#pragma unroll
  for (int off = 1; off < 16; off <<= 1) x += __shfl_xor(x, off, 32);
  return x;
}

// ---------------------------------------------------------------------------
// Pre-pass A: elementwise f32 -> bf16 (8 elements / thread, n % 2048 == 0)
__global__ void __launch_bounds__(256)
mha_cvt_bf16_kernel(const float* __restrict__ in, bf16_t* __restrict__ out, int n) {
  const int i = (blockIdx.x * 256 + threadIdx.x) * 8;
  if (i + 8 <= n) {
    float4 f0 = *(const float4*)(in + i);
    float4 f1 = *(const float4*)(in + i + 4);
    v8bf o;
    o[0]=(bf16_t)f0.x; o[1]=(bf16_t)f0.y; o[2]=(bf16_t)f0.z; o[3]=(bf16_t)f0.w;
    o[4]=(bf16_t)f1.x; o[5]=(bf16_t)f1.y; o[6]=(bf16_t)f1.z; o[7]=(bf16_t)f1.w;
    *(v8bf*)(out + i) = o;
  }
}

// Pre-pass B: W f32 [1024 k][1024 n] -> bf16 [1024 n][1024 k] (LDS 32x32 tiles)
__global__ void __launch_bounds__(256)
mha_transpose_cvt_kernel(const float* __restrict__ in, bf16_t* __restrict__ out) {
  __shared__ bf16_t tile[32][33];
  const int tx = threadIdx.x, ty = threadIdx.y;      // block (32, 8)
  const int n0 = blockIdx.x * 32, k0 = blockIdx.y * 32;
#pragma unroll
  for (int r = 0; r < 4; ++r)
    tile[ty + r * 8][tx] = (bf16_t)in[(size_t)(k0 + ty + r * 8) * DMODEL + n0 + tx];
  __syncthreads();
#pragma unroll
  for (int r = 0; r < 4; ++r)
    out[(size_t)(n0 + ty + r * 8) * DMODEL + k0 + tx] = tile[tx][ty + r * 8];
}

// ---------------------------------------------------------------------------
// Projection GEMM: Y = X(bf16)[8192x1024] @ W  with  Wt = bf16 [N][K].
// grid = (256 Mtiles, 4 Ngroups), block = 128 (4 waves); wave owns 32x64 tile.
// v_transposed: 0 -> write [B,H,S,DH]; 1 -> write [B,H,DH,S] (for V).
__global__ void __launch_bounds__(128)
mha_proj_kernel(const bf16_t* __restrict__ X, const bf16_t* __restrict__ Wt,
                bf16_t* __restrict__ Out, int v_transposed) {
  const int lane = threadIdx.x & 31;
  const int wv   = threadIdx.x >> 5;
  const int half = lane >> 4;
  const int m    = lane & 15;
  const int row0 = blockIdx.x * 32;
  const int nbase = blockIdx.y * 256 + wv * 64;

  v8f acc[2][4] = {};
  const bf16_t* arow0 = X + (size_t)(row0 + m) * DMODEL;
  const bf16_t* arow1 = X + (size_t)(row0 + 16 + m) * DMODEL;
  for (int k0 = 0; k0 < DMODEL; k0 += 32) {
    v16bf a0 = afrag_from_bf16(arow0 + k0, half);
    v16bf a1 = afrag_from_bf16(arow1 + k0, half);
#pragma unroll
    for (int t = 0; t < 4; ++t) {
      const bf16_t* brow = Wt + (size_t)(nbase + t * 16 + m) * DMODEL + k0;
      v16bf b = bfrag_rows_contig(brow, half);
      acc[0][t] = wmma_bf16(a0, b, acc[0][t]);
      acc[1][t] = wmma_bf16(a1, b, acc[1][t]);
    }
  }
#pragma unroll
  for (int t = 0; t < 4; ++t) {
    const int n = nbase + t * 16 + m;
    const int h = n >> 6, dh = n & 63;
#pragma unroll
    for (int u = 0; u < 2; ++u)
#pragma unroll
      for (int v = 0; v < 8; ++v) {
        const int r  = row0 + u * 16 + v + half * 8;   // row in [B*S]
        const int bi = r >> 11, s = r & (SEQ - 1);
        const size_t bhh = (size_t)bi * NH + h;
        const size_t idx = v_transposed ? (bhh * DHEAD + dh) * SEQ + s
                                        : (bhh * SEQ + s) * DHEAD + dh;
        Out[idx] = (bf16_t)acc[u][t][v];
      }
  }
}

// ---------------------------------------------------------------------------
// Causal flash attention. Q,K: [B,H,S,DH] bf16; Vt: [B,H,DH,S] bf16.
// grid = (B*H, SEQ/64), block = 128; each wave owns one 16-query tile.
__global__ void __launch_bounds__(128)
mha_flash_attn_kernel(const bf16_t* __restrict__ Q, const bf16_t* __restrict__ K,
                      const bf16_t* __restrict__ Vt, bf16_t* __restrict__ O_bsd) {
  __shared__ __align__(16) bf16_t pbuf[4][16][40];   // per-wave P re-layout

  const int lane = threadIdx.x & 31;
  const int wv   = threadIdx.x >> 5;
  const int half = lane >> 4;
  const int m    = lane & 15;
  const int bh   = blockIdx.x;
  const int bi   = bh >> 4, h = bh & 15;
  const int q0   = (blockIdx.y * 4 + wv) * 16;

  const bf16_t* Qp  = Q  + (size_t)bh * SEQ * DHEAD;
  const bf16_t* Kp  = K  + (size_t)bh * SEQ * DHEAD;
  const bf16_t* Vtp = Vt + (size_t)bh * DHEAD * SEQ;

  const bf16_t* qrow = Qp + (size_t)(q0 + m) * DHEAD;
  const v16bf aq0 = afrag_from_bf16(qrow, half);        // k = 0..31
  const v16bf aq1 = afrag_from_bf16(qrow + 32, half);   // k = 32..63

  v8f o[4] = {};
  float mrun[8], lrun[8];
#pragma unroll
  for (int v = 0; v < 8; ++v) { mrun[v] = NEG_BIG; lrun[v] = 0.f; }

  bf16_t (*pb)[40] = pbuf[wv];

  const int kend = q0 + 16;                  // causal: keys 0 .. q0+15
  for (int kt0 = 0; kt0 < kend; kt0 += 32) {
    // ---- S = Q K^T for 32 keys (two 16-wide N tiles) ----
    const bf16_t* kr0 = Kp + (size_t)(kt0 + m) * DHEAD;
    const bf16_t* kr1 = Kp + (size_t)(kt0 + 16 + m) * DHEAD;
    v8f s0 = {}, s1 = {};
    s0 = wmma_bf16(aq0, bfrag_rows_contig(kr0, half), s0);
    s0 = wmma_bf16(aq1, bfrag_rows_contig(kr0 + 32, half), s0);
    s1 = wmma_bf16(aq0, bfrag_rows_contig(kr1, half), s1);
    s1 = wmma_bf16(aq1, bfrag_rows_contig(kr1 + 32, half), s1);

    // ---- online softmax; C-tile row = v + half*8, col = lane&15 ----
#pragma unroll
    for (int v = 0; v < 8; ++v) {
      const int qr = q0 + v + half * 8;
      float x0 = (kt0 + m      <= qr) ? s0[v] * ATT_SCALE : NEG_BIG;
      float x1 = (kt0 + 16 + m <= qr) ? s1[v] * ATT_SCALE : NEG_BIG;
      float mnew  = fmaxf(mrun[v], rowmax16(fmaxf(x0, x1)));
      float alpha = __expf(mrun[v] - mnew);
      float p0 = __expf(x0 - mnew);
      float p1 = __expf(x1 - mnew);
      lrun[v] = lrun[v] * alpha + rowsum16(p0 + p1);
      mrun[v] = mnew;
#pragma unroll
      for (int t = 0; t < 4; ++t) o[t][v] *= alpha;
      pb[v + half * 8][m]      = (bf16_t)p0;   // stage P for C->A re-layout
      pb[v + half * 8][16 + m] = (bf16_t)p1;
    }

    // ---- O += P @ V  (P via LDS; V rows contiguous thanks to Vt layout) ----
    v16bf ap = afrag_from_bf16(pb[m], half);   // same-wave LDS RAW: in-order
#pragma unroll
    for (int t = 0; t < 4; ++t) {
      const bf16_t* vrow = Vtp + (size_t)(t * 16 + m) * SEQ + kt0;
      o[t] = wmma_bf16(ap, bfrag_rows_contig(vrow, half), o[t]);
    }
  }

  // ---- normalize and write [B,S,H*DH] bf16 ----
#pragma unroll
  for (int t = 0; t < 4; ++t)
#pragma unroll
    for (int v = 0; v < 8; ++v) {
      const int r = q0 + v + half * 8;
      O_bsd[((size_t)bi * SEQ + r) * DMODEL + h * DHEAD + t * 16 + m] =
          (bf16_t)(o[t][v] / lrun[v]);
    }
}

// ---------------------------------------------------------------------------
// Output projection: out(f32) = A(bf16)[8192x1024] @ Wo, Wot = bf16 [N][K].
__global__ void __launch_bounds__(128)
mha_oproj_kernel(const bf16_t* __restrict__ A, const bf16_t* __restrict__ Wt,
                 float* __restrict__ Out) {
  const int lane = threadIdx.x & 31;
  const int wv   = threadIdx.x >> 5;
  const int half = lane >> 4;
  const int m    = lane & 15;
  const int row0 = blockIdx.x * 32;
  const int nbase = blockIdx.y * 256 + wv * 64;

  v8f acc[2][4] = {};
  const bf16_t* arow0 = A + (size_t)(row0 + m) * DMODEL;
  const bf16_t* arow1 = A + (size_t)(row0 + 16 + m) * DMODEL;
  for (int k0 = 0; k0 < DMODEL; k0 += 32) {
    v16bf a0 = afrag_from_bf16(arow0 + k0, half);
    v16bf a1 = afrag_from_bf16(arow1 + k0, half);
#pragma unroll
    for (int t = 0; t < 4; ++t) {
      const bf16_t* brow = Wt + (size_t)(nbase + t * 16 + m) * DMODEL + k0;
      v16bf b = bfrag_rows_contig(brow, half);
      acc[0][t] = wmma_bf16(a0, b, acc[0][t]);
      acc[1][t] = wmma_bf16(a1, b, acc[1][t]);
    }
  }
#pragma unroll
  for (int t = 0; t < 4; ++t) {
    const int n = nbase + t * 16 + m;
#pragma unroll
    for (int u = 0; u < 2; ++u)
#pragma unroll
      for (int v = 0; v < 8; ++v) {
        const int r = row0 + u * 16 + v + half * 8;
        Out[(size_t)r * DMODEL + n] = acc[u][t][v];
      }
  }
}

// ---------------------------------------------------------------------------
extern "C" void kernel_launch(void* const* d_in, const int* in_sizes, int n_in,
                              void* d_out, int out_size, void* d_ws, size_t ws_size,
                              hipStream_t stream) {
  const float* x_q = (const float*)d_in[0];
  const float* x_k = (const float*)d_in[1];
  const float* x_v = (const float*)d_in[2];
  // d_in[3] = mask (causal; computed analytically in-kernel)
  const float* Wq = (const float*)d_in[4];
  const float* Wk = (const float*)d_in[5];
  const float* Wv = (const float*)d_in[6];
  const float* Wo = (const float*)d_in[7];
  float* out = (float*)d_out;

  const size_t nx = (size_t)BATCH * SEQ * DMODEL;     // 8,388,608
  const size_t nw = (size_t)DMODEL * DMODEL;          // 1,048,576
  bf16_t* Xq  = (bf16_t*)d_ws;                        // bf16 activations
  bf16_t* Xk  = Xq + nx;
  bf16_t* Xv  = Xk + nx;
  bf16_t* Wqt = Xv + nx;                              // bf16 [N][K] weights
  bf16_t* Wkt = Wqt + nw;
  bf16_t* Wvt = Wkt + nw;
  bf16_t* Wot = Wvt + nw;
  bf16_t* Qw  = Wot + nw;                             // [B,H,S,DH]
  bf16_t* Kw  = Qw + nx;
  bf16_t* Vtw = Kw + nx;                              // [B,H,DH,S]
  bf16_t* Aw  = Vtw + nx;                             // attn out [B,S,H*DH]

  dim3 blk128(128), blk256(256);
  // 1) activation conversion
  dim3 gCvt((unsigned)(nx / 2048));
  hipLaunchKernelGGL(mha_cvt_bf16_kernel, gCvt, blk256, 0, stream, x_q, Xq, (int)nx);
  hipLaunchKernelGGL(mha_cvt_bf16_kernel, gCvt, blk256, 0, stream, x_k, Xk, (int)nx);
  hipLaunchKernelGGL(mha_cvt_bf16_kernel, gCvt, blk256, 0, stream, x_v, Xv, (int)nx);
  // 2) weight transpose-conversion
  dim3 gTr(DMODEL / 32, DMODEL / 32), blkTr(32, 8);
  hipLaunchKernelGGL(mha_transpose_cvt_kernel, gTr, blkTr, 0, stream, Wq, Wqt);
  hipLaunchKernelGGL(mha_transpose_cvt_kernel, gTr, blkTr, 0, stream, Wk, Wkt);
  hipLaunchKernelGGL(mha_transpose_cvt_kernel, gTr, blkTr, 0, stream, Wv, Wvt);
  hipLaunchKernelGGL(mha_transpose_cvt_kernel, gTr, blkTr, 0, stream, Wo, Wot);
  // 3) Q/K/V projections
  dim3 gProj((BATCH * SEQ) / 32, DMODEL / 256);       // 256 x 4
  hipLaunchKernelGGL(mha_proj_kernel, gProj, blk128, 0, stream, Xq, Wqt, Qw, 0);
  hipLaunchKernelGGL(mha_proj_kernel, gProj, blk128, 0, stream, Xk, Wkt, Kw, 0);
  hipLaunchKernelGGL(mha_proj_kernel, gProj, blk128, 0, stream, Xv, Wvt, Vtw, 1);
  // 4) attention
  dim3 gAttn(BATCH * NH, SEQ / 64);                   // 64 x 32
  hipLaunchKernelGGL(mha_flash_attn_kernel, gAttn, blk128, 0, stream, Qw, Kw, Vtw, Aw);
  // 5) output projection
  hipLaunchKernelGGL(mha_oproj_kernel, gProj, blk128, 0, stream, Aw, Wot, out);
}
